// MoE_48722109006498
// MI455X (gfx1250) — compile-verified, matrix operand order checked by reading, not run on hardware
//
#include <hip/hip_runtime.h>
#include <hip/hip_bf16.h>
#include <cstdint>

// ---------------- problem constants (from reference) ----------------
constexpr int DIM      = 2048;
constexpr int E        = 32;
constexpr int K_TOP    = 4;
constexpr int INTER    = 1024;
constexpr int SHINTER  = 2 * INTER;   // 2048
constexpr int T        = 2048;        // B*S tokens

// ---------------- WMMA vector types (gfx1250, wave32) ----------------
typedef __attribute__((ext_vector_type(16))) __bf16 v16bf;
typedef __attribute__((ext_vector_type(8)))  float  v8f;

// ---------------- helpers ----------------
__device__ __forceinline__ unsigned short f32_to_bf16(float f) {   // RNE (epilogue only)
    unsigned int u = __float_as_uint(f);
    unsigned int r = (u + 0x7FFFu + ((u >> 16) & 1u)) >> 16;
    return (unsigned short)r;
}
// One v_perm_b32: result = { hi[31:16], lo[31:16] } (truncate-to-bf16 pack)
__device__ __forceinline__ uint32_t pk_bf16(float lo, float hi) {
    return __builtin_amdgcn_perm(__float_as_uint(hi), __float_as_uint(lo), 0x07060302u);
}
__device__ __forceinline__ float silu_fast(float g) {
    // g * v_rcp_f32(1 + exp(-g)) : 4 VALU ops, no IEEE divide chain
    return g * __builtin_amdgcn_rcpf(1.f + __expf(-g));
}
// Per-lane async copy of 16B global -> LDS (ASYNCcnt); drain with s_wait_asynccnt.
__device__ __forceinline__ void async_g2l_b128(const void* gaddr, void* lds_generic) {
    unsigned lds_off = (unsigned)(unsigned long long)lds_generic;  // flat LDS addr[31:0] == LDS offset
    asm volatile("global_load_async_to_lds_b128 %0, %1, off"
                 :: "v"(lds_off), "v"(gaddr) : "memory");
}
__device__ __forceinline__ void wait_asynccnt0() {
    asm volatile("s_wait_asynccnt 0x0" ::: "memory");
}

// ---------------- small setup kernels ----------------
__global__ void init_counts_kernel(int* counts) {
    if (threadIdx.x < E) counts[threadIdx.x] = 0;
}

__global__ void cvt_bf16_kernel(const float* __restrict__ x,
                                unsigned short* __restrict__ xb, int n) {
    int i = blockIdx.x * blockDim.x + threadIdx.x;
    if (i < n) xb[i] = f32_to_bf16(x[i]);
}

// One wave32 per token; lane == expert. Softmax + top-4 inside the wave.
__global__ void gate_kernel(const float* __restrict__ x,
                            const float* __restrict__ gw,
                            int* __restrict__ topi, float* __restrict__ topw,
                            int* __restrict__ counts) {
    const int t    = blockIdx.x;
    const int lane = threadIdx.x;           // 0..31 == expert id
    const float* xr = x  + (size_t)t    * DIM;
    const float* gr = gw + (size_t)lane * DIM;
    float acc = 0.f;
    for (int d = 0; d < DIM; d += 4) {
        float4 xv = *(const float4*)(xr + d);
        float4 gv = *(const float4*)(gr + d);
        acc += xv.x * gv.x + xv.y * gv.y + xv.z * gv.z + xv.w * gv.w;
    }
    float m = acc;
    for (int o = 16; o > 0; o >>= 1) m = fmaxf(m, __shfl_xor(m, o, 32));
    float e = __expf(acc - m);
    float s = e;
    for (int o = 16; o > 0; o >>= 1) s += __shfl_xor(s, o, 32);
    float v = e / s;
    for (int k = 0; k < K_TOP; ++k) {
        float mx = v;
        for (int o = 16; o > 0; o >>= 1) mx = fmaxf(mx, __shfl_xor(mx, o, 32));
        int sel = (v == mx) ? lane : 64;
        for (int o = 16; o > 0; o >>= 1) sel = min(sel, __shfl_xor(sel, o, 32));
        if (lane == 0) {
            topi[t * K_TOP + k] = sel;
            topw[t * K_TOP + k] = mx;
            atomicAdd(&counts[sel], 1);
        }
        if (lane == sel) v = -1.f;          // probs >= 0, acts as -inf
    }
}

__global__ void scan_kernel(const int* __restrict__ counts,
                            int* __restrict__ offs, int* __restrict__ cursor) {
    if (threadIdx.x == 0) {
        int s = 0;
        for (int e = 0; e < E; ++e) { offs[e] = s; cursor[e] = s; s += counts[e]; }
    }
}

__global__ void scatter_kernel(const int* __restrict__ topi,
                               const float* __restrict__ topw,
                               int* __restrict__ cursor,
                               int* __restrict__ tok, float* __restrict__ wgt) {
    int t = blockIdx.x * blockDim.x + threadIdx.x;
    if (t >= T) return;
    for (int k = 0; k < K_TOP; ++k) {
        int e   = topi[t * K_TOP + k];
        int pos = atomicAdd(&cursor[e], 1);
        tok[pos] = t;
        wgt[pos] = topw[t * K_TOP + k];
    }
}

// ---------------- WMMA GEMM tiles ----------------
// 256 threads = 8 waves.  Block tile: BM=64 x BN=64, K-step 32.
// Waves: wm = wid&1 (row-group), wn = wid>>1 (4 col-strips).
// Each wave owns TWO 16-row M-subtiles (rows wm*16 and 32+wm*16), reusing one
// B fragment for 2x the WMMAs -> halves per-WMMA weight fetch cost.
constexpr int BM = 64;
constexpr int BN = 64;
constexpr int SLOTS = T * K_TOP;          // routed h rows (dump rows live past this)

union FragBF { v16bf v; uint32_t u[8]; };

// h = silu(X @ W1^T) * (X @ W3^T), bf16 output.
template <bool GATHER, int NINTER>
__global__ __launch_bounds__(256) void up_kernel(
    const unsigned short* __restrict__ xb,
    const float* __restrict__ W1b, const float* __restrict__ W3b, size_t wstride,
    unsigned short* __restrict__ hout,
    const int* __restrict__ offs, const int* __restrict__ counts,
    const int* __restrict__ tok, int dense_rows)
{
    const int e = blockIdx.y;
    int cnt, off;
    if (GATHER) { cnt = counts[e]; off = offs[e]; }
    else        { cnt = dense_rows; off = 0; }
    const int m0 = blockIdx.x * BM;
    if (m0 >= cnt) return;
    const int nb = blockIdx.z * BN;
    const float* W1 = W1b + (size_t)e * wstride;
    const float* W3 = W3b + (size_t)e * wstride;

    __shared__ unsigned short sA[BM][40];   // 64 x K32 bf16 tile (+pad)
    __shared__ int sTok[BM];

    const int tid = threadIdx.x;
    if (tid < BM) {
        int r = m0 + tid;
        int rr = (r < cnt) ? r : (cnt - 1);
        sTok[tid] = GATHER ? tok[off + rr] : rr;
    }
    __syncthreads();

    const int lane = tid & 31, wid = tid >> 5;
    const int wm = wid & 1, wn = wid >> 1;

    v8f a1[2] = {v8f{}, v8f{}};
    v8f a3[2] = {v8f{}, v8f{}};

    const int ld_row = tid >> 2;            // 64 rows, 4 threads/row
    const int ld_col = (tid & 3) * 8;       // 8 bf16 = 16B per thread per step
    const unsigned short* asrc = xb + (size_t)sTok[ld_row] * DIM + ld_col;
    unsigned short* adst = &sA[ld_row][ld_col];

    const int bn_lane = nb + wn * 16 + (lane & 15);
    const float* p1 = W1 + (size_t)bn_lane * DIM + (lane >> 4) * 16;
    const float* p3 = W3 + (size_t)bn_lane * DIM + (lane >> 4) * 16;
    const unsigned short* arow0 = &sA[wm * 16 + (lane & 15)][(lane >> 4) * 16];
    const unsigned short* arow1 = &sA[32 + wm * 16 + (lane & 15)][(lane >> 4) * 16];

    for (int k = 0; k < DIM; k += 32) {
        __syncthreads();                      // previous tile consumed
        async_g2l_b128(asrc + k, adst);       // gfx1250 async copy global->LDS
        wait_asynccnt0();                     // drain this wave's async ops
        __syncthreads();                      // publish all waves' LDS writes

        FragBF fb1, fb3;
#pragma unroll
        for (int i = 0; i < 8; ++i) {
            fb1.u[i] = pk_bf16(p1[k + 2 * i], p1[k + 2 * i + 1]);   // 1 v_perm each
            fb3.u[i] = pk_bf16(p3[k + 2 * i], p3[k + 2 * i + 1]);
        }
        __builtin_prefetch(p1 + k + 128, 0, 1);   // global_prefetch_b8
        __builtin_prefetch(p3 + k + 128, 0, 1);

        FragBF fa0, fa1v;
#pragma unroll
        for (int i = 0; i < 8; ++i) fa0.u[i]  = ((const uint32_t*)arow0)[i];
#pragma unroll
        for (int i = 0; i < 8; ++i) fa1v.u[i] = ((const uint32_t*)arow1)[i];

        a1[0] = __builtin_amdgcn_wmma_f32_16x16x32_bf16(false, fa0.v,  false, fb1.v, (short)0, a1[0], false, false);
        a3[0] = __builtin_amdgcn_wmma_f32_16x16x32_bf16(false, fa0.v,  false, fb3.v, (short)0, a3[0], false, false);
        a1[1] = __builtin_amdgcn_wmma_f32_16x16x32_bf16(false, fa1v.v, false, fb1.v, (short)0, a1[1], false, false);
        a3[1] = __builtin_amdgcn_wmma_f32_16x16x32_bf16(false, fa1v.v, false, fb3.v, (short)0, a3[1], false, false);
    }

    const int colg = nb + wn * 16 + (lane & 15);
#pragma unroll
    for (int mt = 0; mt < 2; ++mt) {
#pragma unroll
        for (int i = 0; i < 8; ++i) {
            int r = mt * 32 + wm * 16 + ((lane >> 4) << 3) + i;  // C layout: M = i (+8 hi lanes)
            float h = silu_fast(a1[mt][i]) * a3[mt][i];
            size_t slot;
            if (GATHER) // branch-free: invalid rows land in dump rows past SLOTS
                slot = (m0 + r < cnt) ? (size_t)(off + m0 + r) : (size_t)(SLOTS + r);
            else        // dense rows: grid covers exactly, always valid
                slot = (size_t)(m0 + r);
            hout[slot * NINTER + colg] = f32_to_bf16(h);
        }
    }
}

// y = H @ W2^T.  SCATTER: atomicAdd(out[tok[row]], y*wgt[row]); else plain store.
template <bool SCATTER, int KDIM>
__global__ __launch_bounds__(256) void down_kernel(
    const unsigned short* __restrict__ hin,
    const float* __restrict__ W2b, size_t wstride,
    float* __restrict__ out,
    const int* __restrict__ offs, const int* __restrict__ counts,
    const int* __restrict__ tok, const float* __restrict__ wgt, int dense_rows)
{
    const int e = blockIdx.y;
    int cnt, off;
    if (SCATTER) { cnt = counts[e]; off = offs[e]; }
    else         { cnt = dense_rows; off = 0; }
    const int m0 = blockIdx.x * BM;
    if (m0 >= cnt) return;
    const int nb = blockIdx.z * BN;
    const float* W2 = W2b + (size_t)e * wstride;

    __shared__ unsigned short sA[BM][40];
    __shared__ int   sTok[BM];
    __shared__ float sW[BM];

    const int tid = threadIdx.x;
    if (SCATTER && tid < BM) {
        int r = m0 + tid;
        int rr = (r < cnt) ? r : (cnt - 1);
        sTok[tid] = tok[off + rr];
        sW[tid]   = (r < cnt) ? wgt[off + rr] : 0.f;   // zero-weight masks overhang rows
    }
    __syncthreads();

    const int lane = tid & 31, wid = tid >> 5;
    const int wm = wid & 1, wn = wid >> 1;
    v8f acc[2] = {v8f{}, v8f{}};

    const int ld_row = tid >> 2;
    const int ld_col = (tid & 3) * 8;
    // slot rows are contiguous per expert -> no gather needed on the A side
    const unsigned short* asrc = hin + (size_t)(off + m0 + ld_row) * KDIM + ld_col;
    unsigned short* adst = &sA[ld_row][ld_col];

    const int bn_lane = nb + wn * 16 + (lane & 15);
    const float* p2 = W2 + (size_t)bn_lane * KDIM + (lane >> 4) * 16;
    const unsigned short* arow0 = &sA[wm * 16 + (lane & 15)][(lane >> 4) * 16];
    const unsigned short* arow1 = &sA[32 + wm * 16 + (lane & 15)][(lane >> 4) * 16];

    for (int k = 0; k < KDIM; k += 32) {
        __syncthreads();
        async_g2l_b128(asrc + k, adst);
        wait_asynccnt0();
        __syncthreads();

        FragBF fb;
#pragma unroll
        for (int i = 0; i < 8; ++i)
            fb.u[i] = pk_bf16(p2[k + 2 * i], p2[k + 2 * i + 1]);
        __builtin_prefetch(p2 + k + 128, 0, 1);

        FragBF fa0, fa1v;
#pragma unroll
        for (int i = 0; i < 8; ++i) fa0.u[i]  = ((const uint32_t*)arow0)[i];
#pragma unroll
        for (int i = 0; i < 8; ++i) fa1v.u[i] = ((const uint32_t*)arow1)[i];

        acc[0] = __builtin_amdgcn_wmma_f32_16x16x32_bf16(false, fa0.v,  false, fb.v, (short)0, acc[0], false, false);
        acc[1] = __builtin_amdgcn_wmma_f32_16x16x32_bf16(false, fa1v.v, false, fb.v, (short)0, acc[1], false, false);
    }

    const int colg = nb + wn * 16 + (lane & 15);
#pragma unroll
    for (int mt = 0; mt < 2; ++mt) {
#pragma unroll
        for (int i = 0; i < 8; ++i) {
            int r = mt * 32 + wm * 16 + ((lane >> 4) << 3) + i;
            float v = acc[mt][i];
            if (SCATTER) {
                // branch-free: overhang rows have weight 0 -> adds 0 to a valid token
                atomicAdd(&out[(size_t)sTok[r] * DIM + colg], v * sW[r]);
            } else {
                out[(size_t)(m0 + r) * DIM + colg] = v;
            }
        }
    }
}

// ---------------- launcher ----------------
extern "C" void kernel_launch(void* const* d_in, const int* in_sizes, int n_in,
                              void* d_out, int out_size, void* d_ws, size_t ws_size,
                              hipStream_t stream) {
    (void)in_sizes; (void)n_in; (void)out_size; (void)ws_size;

    const float* x      = (const float*)d_in[0];
    const float* gate_w = (const float*)d_in[1];
    const float* w1     = (const float*)d_in[2];
    const float* w2     = (const float*)d_in[3];
    const float* w3     = (const float*)d_in[4];
    const float* sw1    = (const float*)d_in[5];
    const float* sw2    = (const float*)d_in[6];
    const float* sw3    = (const float*)d_in[7];
    float* out = (float*)d_out;

    char* ws = (char*)d_ws;
    size_t o = 0;
    auto alloc = [&](size_t bytes) -> char* {
        o = (o + 255) & ~(size_t)255;
        char* p = ws + o;
        o += bytes;
        return p;
    };

    unsigned short* xb      = (unsigned short*)alloc((size_t)T * DIM * 2);
    int*            counts  = (int*)  alloc(E * sizeof(int));
    int*            offs    = (int*)  alloc(E * sizeof(int));
    int*            cursor  = (int*)  alloc(E * sizeof(int));
    int*            topi    = (int*)  alloc((size_t)T * K_TOP * sizeof(int));
    float*          topw    = (float*)alloc((size_t)T * K_TOP * sizeof(float));
    int*            tokl    = (int*)  alloc((size_t)T * K_TOP * sizeof(int));
    float*          wgtl    = (float*)alloc((size_t)T * K_TOP * sizeof(float));
    unsigned short* hroute  = (unsigned short*)alloc(((size_t)SLOTS + BM) * INTER * 2);
    unsigned short* hshared = (unsigned short*)alloc((size_t)T * SHINTER * 2);

    // 1) routing metadata
    init_counts_kernel<<<1, 32, 0, stream>>>(counts);
    cvt_bf16_kernel<<<(T * DIM + 255) / 256, 256, 0, stream>>>(x, xb, T * DIM);
    gate_kernel<<<T, 32, 0, stream>>>(x, gate_w, topi, topw, counts);
    scan_kernel<<<1, 32, 0, stream>>>(counts, offs, cursor);
    scatter_kernel<<<(T + 255) / 256, 256, 0, stream>>>(topi, topw, cursor, tokl, wgtl);

    // 2) shared expert: z = (silu(x sw1^T) * (x sw3^T)) sw2^T  -> initializes out
    up_kernel<false, SHINTER><<<dim3(T / BM, 1, SHINTER / BN), 256, 0, stream>>>(
        xb, sw1, sw3, 0, hshared, nullptr, nullptr, nullptr, T);
    down_kernel<false, SHINTER><<<dim3(T / BM, 1, DIM / BN), 256, 0, stream>>>(
        hshared, sw2, 0, out, nullptr, nullptr, nullptr, nullptr, T);

    // 3) routed experts (gather -> SwiGLU -> weighted scatter-add)
    up_kernel<true, INTER><<<dim3(T / BM, E, INTER / BN), 256, 0, stream>>>(
        xb, w1, w3, (size_t)INTER * DIM, hroute, offs, counts, tokl, 0);
    down_kernel<true, INTER><<<dim3(T / BM, E, DIM / BN), 256, 0, stream>>>(
        hroute, w2, (size_t)DIM * INTER, out, offs, counts, tokl, wgtl, 0);
}
